// SetAbstraction_22522808500492
// MI455X (gfx1250) — matrix operand order repaired and checked by reading, not compiled, not run on hardware
//
#include <hip/hip_runtime.h>

#define BATCH 4
#define NPTS  8192
#define NCENT 1024
#define KNBR  32
#define DFEAT 13
#define RAD2  0.04f      // 0.2^2
#define EPSBN 1e-5f

typedef __attribute__((ext_vector_type(16))) _Float16 v16h;
typedef __attribute__((ext_vector_type(8)))  _Float16 v8h;
typedef __attribute__((ext_vector_type(8)))  float    v8f;

// ---------------------------------------------------------------------------
// 1) Farthest point sampling: one 1024-thread block per batch, 8 points/thread
//    kept in registers; LDS tree argmax per iteration. far0 := 0.
// ---------------------------------------------------------------------------
__global__ __launch_bounds__(1024) void fps_kernel(const float* __restrict__ coords,
                                                   float* __restrict__ cent)
{
    const int b = blockIdx.x, tid = threadIdx.x;
    const float* cb = coords + (size_t)b * NPTS * 3;

    float px[8], py[8], pz[8], dmin[8];
#pragma unroll
    for (int i = 0; i < 8; ++i) {
        int p = tid * 8 + i;
        px[i] = cb[p * 3 + 0];
        py[i] = cb[p * 3 + 1];
        pz[i] = cb[p * 3 + 2];
        dmin[i] = 3.0e38f;
    }

    __shared__ float sval[1024];
    __shared__ int   sidx[1024];
    __shared__ float scen[3];

    int far = 0;
    for (int c = 0; c < NCENT; ++c) {
        if ((far >> 3) == tid) {                  // owner broadcasts centroid
            int i = far & 7;
            scen[0] = px[i]; scen[1] = py[i]; scen[2] = pz[i];
            float* co = cent + ((size_t)b * NCENT + c) * 3;
            co[0] = px[i]; co[1] = py[i]; co[2] = pz[i];
        }
        __syncthreads();
        const float cx = scen[0], cy = scen[1], cz = scen[2];

        float best = -1.0f; int bi = tid * 8;
#pragma unroll
        for (int i = 0; i < 8; ++i) {
            float dx = px[i] - cx, dy = py[i] - cy, dz = pz[i] - cz;
            float d2 = dx * dx + dy * dy + dz * dz;   // argmax(min d) == argmax(min d^2)
            dmin[i] = fminf(dmin[i], d2);
            if (dmin[i] > best) { best = dmin[i]; bi = tid * 8 + i; }
        }
        sval[tid] = best; sidx[tid] = bi;
        __syncthreads();
        for (int s = 512; s > 0; s >>= 1) {           // first-index-on-tie argmax
            if (tid < s) {
                float ov = sval[tid + s]; int oi = sidx[tid + s];
                if (ov > sval[tid] || (ov == sval[tid] && oi < sidx[tid])) {
                    sval[tid] = ov; sidx[tid] = oi;
                }
            }
            __syncthreads();
        }
        far = sidx[0];   // next iteration's top barrier orders this vs. overwrite
    }
}

// ---------------------------------------------------------------------------
// 2) Ball query + top-K grouping. 128 centroids per block (1 thread each),
//    coord tiles in LDS (broadcast), per-thread sorted top-32 list in LDS
//    (stride 33 to dodge bank conflicts). Emits f16 rows zero-padded to 32.
// ---------------------------------------------------------------------------
#define GT 128
#define TP 1024
#define LSTRIDE 33

__global__ __launch_bounds__(GT) void group_kernel(const float* __restrict__ coords,
                                                   const float* __restrict__ feats,
                                                   const float* __restrict__ cent,
                                                   _Float16* __restrict__ x0)
{
    __shared__ float sx[TP], sy[TP], sz[TP];
    __shared__ float ld[GT * LSTRIDE];
    __shared__ int   li[GT * LSTRIDE];

    const int tid = threadIdx.x;
    const int cpb = NCENT / GT;
    const int b = blockIdx.x / cpb;
    const int c = (blockIdx.x % cpb) * GT + tid;

    const float* cb = coords + (size_t)b * NPTS * 3;
    const float* cc = cent + ((size_t)b * NCENT + c) * 3;
    const float ccx = cc[0], ccy = cc[1], ccz = cc[2];
    const int base = tid * LSTRIDE;

    int cnt = 0;
    for (int t = 0; t < NPTS / TP; ++t) {
        for (int j = tid; j < TP; j += GT) {
            int p = t * TP + j;
            sx[j] = cb[p * 3 + 0]; sy[j] = cb[p * 3 + 1]; sz[j] = cb[p * 3 + 2];
        }
        __syncthreads();
        for (int j = 0; j < TP; ++j) {
            float dx = sx[j] - ccx, dy = sy[j] - ccy, dz = sz[j] - ccz;
            float d2 = dx * dx + dy * dy + dz * dz;
            if (d2 <= RAD2) {
                if (cnt < KNBR) {
                    int pos = cnt;
                    while (pos > 0 && ld[base + pos - 1] > d2) {
                        ld[base + pos] = ld[base + pos - 1];
                        li[base + pos] = li[base + pos - 1];
                        --pos;
                    }
                    ld[base + pos] = d2; li[base + pos] = t * TP + j; ++cnt;
                } else if (d2 < ld[base + KNBR - 1]) {
                    int pos = KNBR - 1;
                    while (pos > 0 && ld[base + pos - 1] > d2) {
                        ld[base + pos] = ld[base + pos - 1];
                        li[base + pos] = li[base + pos - 1];
                        --pos;
                    }
                    ld[base + pos] = d2; li[base + pos] = t * TP + j;
                }
            }
        }
        __syncthreads();
    }

    const int pad = (cnt > 0) ? li[base] : 0;
    _Float16* xr = x0 + ((size_t)(b * NCENT + c) * KNBR) * 32;
    for (int k = 0; k < KNBR; ++k) {
        int idx = (k < cnt) ? li[base + k] : pad;
        const float* pc = cb + (size_t)idx * 3;
        const float* pf = feats + ((size_t)b * NPTS + idx) * DFEAT;
        _Float16* row = xr + (size_t)k * 32;
        row[0] = (_Float16)(pc[0] - ccx);
        row[1] = (_Float16)(pc[1] - ccy);
        row[2] = (_Float16)(pc[2] - ccz);
#pragma unroll
        for (int d = 0; d < DFEAT; ++d) row[3 + d] = (_Float16)pf[d];
#pragma unroll
        for (int d = 16; d < 32; ++d) row[d] = (_Float16)0.f;
    }
}

// ---------------------------------------------------------------------------
// 3) Weight-stationary WMMA GEMM + bias + per-channel sum/sumsq accumulation.
//    One wave per 16-channel output tile. Compile-time KP/FOUT -> branch-free
//    fully unrolled inner loop; A loads for m-tile mt+1 are issued before the
//    WMMAs consuming m-tile mt (software pipeline keeps loads in flight under
//    the matrix ops). A/B operands are contiguous vector loads per the CDNA5
//    16-bit WMMA VGPR layouts.
// ---------------------------------------------------------------------------
template<int KP, int FOUT, int MT>
__global__ void gemm_bn_stats_kernel(const _Float16* __restrict__ X,
                                     const _Float16* __restrict__ W,   // [FOUT][K] f16 (K padded)
                                     const float* __restrict__ bias,
                                     _Float16* __restrict__ Y,
                                     float* __restrict__ gsum, float* __restrict__ gsq)
{
    constexpr int K = KP * 32;
    const int lane = threadIdx.x & 31;
    const int nt   = threadIdx.x >> 5;      // 16-col output tile index
    const int nl   = lane & 15;
    const int hi   = lane >> 4;             // lane-group: 0 -> k 0-7,16-23 of each 32-k step, 1 -> k 8-15,24-31
    const int col  = nt * 16 + nl;

    // B operand (weights stationary): lane holds K-values hi*8..hi*8+7 and 16+hi*8..16+hi*8+7 of row `col`
    const _Float16* wrow = W + (size_t)col * K;
    v16h breg[KP];
#pragma unroll
    for (int kp = 0; kp < KP; ++kp) {
        v8h b0 = *(const v8h*)(wrow + kp * 32 + hi * 8);
        v8h b1 = *(const v8h*)(wrow + kp * 32 + hi * 8 + 16);
        breg[kp] = __builtin_shufflevector(b0, b1, 0, 1, 2, 3, 4, 5, 6, 7, 8, 9, 10, 11, 12, 13, 14, 15);
    }

    const float bc = bias[col];
    float ssum = 0.f, ssq = 0.f;

    union Au { v16h v; v8h h[2]; };
    Au cur[KP], nxt[KP];

    const int rowblk = blockIdx.x * 16 * MT;
    const _Float16* abase = X + (size_t)(rowblk + nl) * K + hi * 8;

    // prologue: load m-tile 0
#pragma unroll
    for (int kp = 0; kp < KP; ++kp) {
        cur[kp].h[0] = *(const v8h*)(abase + kp * 32);
        cur[kp].h[1] = *(const v8h*)(abase + kp * 32 + 16);
    }

#pragma unroll
    for (int mt = 0; mt < MT; ++mt) {
        // issue next m-tile's loads before consuming the current one
        if (mt + 1 < MT) {
            const _Float16* ap = abase + (size_t)(mt + 1) * 16 * K;
#pragma unroll
            for (int kp = 0; kp < KP; ++kp) {
                nxt[kp].h[0] = *(const v8h*)(ap + kp * 32);
                nxt[kp].h[1] = *(const v8h*)(ap + kp * 32 + 16);
            }
        } else {
            // prefetch the next block-row of A (L2-resident stream)
            __builtin_prefetch(abase + (size_t)MT * 16 * K, 0, 1);
        }

        v8f acc = {};
#pragma unroll
        for (int kp = 0; kp < KP; ++kp)
            acc = __builtin_amdgcn_wmma_f32_16x16x32_f16(false, cur[kp].v, false, breg[kp],
                                                         (short)0, acc, false, false);

        const int row0 = rowblk + mt * 16;
#pragma unroll
        for (int i = 0; i < 8; ++i) {       // C layout: row = i + 8*hi, col = nl
            float v = acc[i] + bc;
            ssum += v; ssq += v * v;
            Y[(size_t)(row0 + i + hi * 8) * FOUT + col] = (_Float16)v;
        }

#pragma unroll
        for (int kp = 0; kp < KP; ++kp) cur[kp] = nxt[kp];
    }

    ssum += __shfl_xor(ssum, 16);           // lanes L and L^16 share a column
    ssq  += __shfl_xor(ssq, 16);
    if (hi == 0) {
        atomicAdd(&gsum[col], ssum);
        atomicAdd(&gsq[col], ssq);
    }
}

// ---------------------------------------------------------------------------
// 4) BN finalize: fold (mu, var, gamma, beta) into per-channel scale/shift.
// ---------------------------------------------------------------------------
__global__ void bn_finalize_kernel(const float* __restrict__ gsum, const float* __restrict__ gsq,
                                   const float* __restrict__ g, const float* __restrict__ beta,
                                   float* __restrict__ sc, float* __restrict__ sh,
                                   int Fout, float invM)
{
    int c = threadIdx.x;
    if (c < Fout) {
        float mu  = gsum[c] * invM;
        float var = gsq[c] * invM - mu * mu;
        float s   = g[c] * rsqrtf(var + EPSBN);
        sc[c] = s;
        sh[c] = beta[c] - mu * s;
    }
}

// ---------------------------------------------------------------------------
// 5) In-place normalize + ReLU (Fout is a power of two).
// ---------------------------------------------------------------------------
__global__ void bn_relu_kernel(_Float16* __restrict__ Y,
                               const float* __restrict__ sc, const float* __restrict__ sh,
                               size_t total, int Fout)
{
    size_t i = (size_t)blockIdx.x * blockDim.x + threadIdx.x;
    size_t stride = (size_t)gridDim.x * blockDim.x;
    for (; i < total; i += stride) {
        int c = (int)(i & (size_t)(Fout - 1));
        float v = (float)Y[i] * sc[c] + sh[c];
        Y[i] = (_Float16)fmaxf(v, 0.f);
    }
}

// ---------------------------------------------------------------------------
// 6) Fused normalize + ReLU + max pool over K for the final layer.
// ---------------------------------------------------------------------------
__global__ __launch_bounds__(128) void pool_kernel(const _Float16* __restrict__ Y3,
                                                   const float* __restrict__ sc,
                                                   const float* __restrict__ sh,
                                                   float* __restrict__ out)
{
    const int bc = blockIdx.x, ch = threadIdx.x;
    const float s = sc[ch], t = sh[ch];
    float mx = 0.f;                                   // ReLU outputs are >= 0
    const _Float16* yp = Y3 + (size_t)bc * KNBR * 128 + ch;
#pragma unroll
    for (int k = 0; k < KNBR; ++k) {
        float v = (float)yp[(size_t)k * 128] * s + t;
        mx = fmaxf(mx, fmaxf(v, 0.f));
    }
    out[(size_t)bc * 128 + ch] = mx;
}

// ---------------------------------------------------------------------------
// 7) Convert weights f32 -> f16 (w0 zero-padded 16 -> 32 along K).
// ---------------------------------------------------------------------------
__global__ void cvt_weights_kernel(const float* __restrict__ w0, const float* __restrict__ w1,
                                   const float* __restrict__ w2,
                                   _Float16* __restrict__ w0p, _Float16* __restrict__ w1p,
                                   _Float16* __restrict__ w2p)
{
    int i = blockIdx.x * blockDim.x + threadIdx.x;
    if (i < 64 * 32) {
        int r = i >> 5, c = i & 31;
        w0p[i] = (_Float16)(c < 16 ? w0[r * 16 + c] : 0.f);
    }
    if (i < 64 * 64)  w1p[i] = (_Float16)w1[i];
    if (i < 128 * 64) w2p[i] = (_Float16)w2[i];
}

// ---------------------------------------------------------------------------
extern "C" void kernel_launch(void* const* d_in, const int* in_sizes, int n_in,
                              void* d_out, int out_size, void* d_ws, size_t ws_size,
                              hipStream_t stream)
{
    (void)in_sizes; (void)n_in; (void)out_size; (void)ws_size;

    const float* coords = (const float*)d_in[0];
    const float* feats  = (const float*)d_in[1];
    const float* w0  = (const float*)d_in[2];
    const float* b0  = (const float*)d_in[3];
    const float* g0  = (const float*)d_in[4];
    const float* be0 = (const float*)d_in[5];
    const float* w1  = (const float*)d_in[6];
    const float* b1  = (const float*)d_in[7];
    const float* g1  = (const float*)d_in[8];
    const float* be1 = (const float*)d_in[9];
    const float* w2  = (const float*)d_in[10];
    const float* b2  = (const float*)d_in[11];
    const float* g2  = (const float*)d_in[12];
    const float* be2 = (const float*)d_in[13];

    float* out    = (float*)d_out;
    float* cent   = out;                               // (B, NC, 3)
    float* pooled = out + (size_t)BATCH * NCENT * 3;   // (B, NC, 128)

    const size_t M = (size_t)BATCH * NCENT * KNBR;     // 131072 rows

    char* ws = (char*)d_ws;
    size_t o = 0;
    auto take = [&](size_t bytes) -> void* {
        void* p = ws + o;
        o += (bytes + 255) & ~(size_t)255;
        return p;
    };
    _Float16* x0  = (_Float16*)take(M * 32  * sizeof(_Float16));
    _Float16* y1  = (_Float16*)take(M * 64  * sizeof(_Float16));
    _Float16* y2  = (_Float16*)take(M * 64  * sizeof(_Float16));
    _Float16* y3  = (_Float16*)take(M * 128 * sizeof(_Float16));
    _Float16* w0p = (_Float16*)take(64 * 32  * sizeof(_Float16));
    _Float16* w1p = (_Float16*)take(64 * 64  * sizeof(_Float16));
    _Float16* w2p = (_Float16*)take(128 * 64 * sizeof(_Float16));
    float* st     = (float*)take(3 * 512 * sizeof(float)); // per layer: sum,sq,sc,sh (128 each)

    hipMemsetAsync(st, 0, 3 * 512 * sizeof(float), stream); // zero atomics accumulators

    cvt_weights_kernel<<<32, 256, 0, stream>>>(w0, w1, w2, w0p, w1p, w2p);
    fps_kernel<<<BATCH, 1024, 0, stream>>>(coords, cent);
    group_kernel<<<BATCH * NCENT / GT, GT, 0, stream>>>(coords, feats, cent, x0);

    constexpr int MT = 8;
    const int mblocks = (int)(M / (16 * MT));          // 1024
    const float invM = 1.0f / (float)M;

    // layer 0: 16(pad 32) -> 64
    gemm_bn_stats_kernel<1, 64, MT><<<mblocks, 128, 0, stream>>>(x0, w0p, b0, y1,
                                                                 st + 0, st + 128);
    bn_finalize_kernel<<<1, 64, 0, stream>>>(st + 0, st + 128, g0, be0,
                                             st + 256, st + 384, 64, invM);
    bn_relu_kernel<<<2048, 256, 0, stream>>>(y1, st + 256, st + 384, M * 64, 64);

    // layer 1: 64 -> 64
    gemm_bn_stats_kernel<2, 64, MT><<<mblocks, 128, 0, stream>>>(y1, w1p, b1, y2,
                                                                 st + 512, st + 640);
    bn_finalize_kernel<<<1, 64, 0, stream>>>(st + 512, st + 640, g1, be1,
                                             st + 768, st + 896, 64, invM);
    bn_relu_kernel<<<2048, 256, 0, stream>>>(y2, st + 768, st + 896, M * 64, 64);

    // layer 2: 64 -> 128, fused BN+ReLU+maxpool in pool_kernel
    gemm_bn_stats_kernel<2, 128, MT><<<mblocks, 256, 0, stream>>>(y2, w2p, b2, y3,
                                                                  st + 1024, st + 1152);
    bn_finalize_kernel<<<1, 128, 0, stream>>>(st + 1024, st + 1152, g2, be2,
                                              st + 1280, st + 1408, 128, invM);
    pool_kernel<<<BATCH * NCENT, 128, 0, stream>>>(y3, st + 1280, st + 1408, pooled);
}
